// GraphClassifier_18038862643346
// MI455X (gfx1250) — matrix operand order repaired and checked by reading, not compiled, not run on hardware
//
#include <hip/hip_runtime.h>
#include <hip/hip_bf16.h>

// ---------------------------------------------------------------------------
// CDNA5 (gfx1250) types
// ---------------------------------------------------------------------------
typedef __bf16 bf16_t;
typedef __bf16 v16bf __attribute__((ext_vector_type(16)));
typedef float  v8f   __attribute__((ext_vector_type(8)));
typedef float  vf4   __attribute__((ext_vector_type(4)));
typedef int    vi4   __attribute__((ext_vector_type(4)));
typedef unsigned int vu2 __attribute__((ext_vector_type(2)));

// pack 4 f32 -> 4 bf16 (v_cvt_pk_bf16_f32 x2) as 8 bytes
__device__ __forceinline__ vu2 pack4_bf16(vf4 v) {
    unsigned h0 = __builtin_bit_cast(unsigned short, (bf16_t)v.x);
    unsigned h1 = __builtin_bit_cast(unsigned short, (bf16_t)v.y);
    unsigned h2 = __builtin_bit_cast(unsigned short, (bf16_t)v.z);
    unsigned h3 = __builtin_bit_cast(unsigned short, (bf16_t)v.w);
    vu2 r; r.x = h0 | (h1 << 16); r.y = h2 | (h3 << 16);
    return r;
}

// Fragment-ready LDS placement (inverse of documented 16-bit WMMA layouts):
//   A(16x32): lane = ((k>>3)&1)*16 + m ; pos = (k&7) | ((k>>4)<<3)
//   B(32x16): lane = ((k>>3)&1)*16 + n ; pos = (k&7) | ((k>>4)<<3)
// => each lane's fragment = 16 contiguous bf16 (32B) -> 2x ds_load_b128.

// ---------------------------------------------------------------------------
// GEMM: C[M,N] = A[M,K] @ Wt[N,K]^T + bias.  Block = 32 rows x N cols,
// 256 threads (8 waves): wave w -> row tile (w>>2), col tiles (w&3)*TPW..+TPW-1.
// N == TPW*4*16. A is read exactly ONCE from memory (full-N blocks).
// Double-buffered LDS; fragment ds_loads issued BEFORE next-tile prefetch so
// the WMMA only waits on the loads, overlapping prefetch with the matrix op.
// ---------------------------------------------------------------------------
template<int TPW, bool NTA>
__global__ __launch_bounds__(256)
void gemm_wmma(const float* __restrict__ A,
               const float* __restrict__ Wt,
               const float* __restrict__ bias,
               float* __restrict__ C,
               unsigned M, unsigned N, unsigned K)
{
    constexpr int NT = TPW * 4;                       // col tiles per block
    __shared__ __align__(32) bf16_t Af[2][2][32][16];
    __shared__ __align__(32) bf16_t Bf[2][NT][32][16];

    const unsigned tid  = threadIdx.x;
    const unsigned lane = tid & 31u, wave = tid >> 5;
    const unsigned half = lane >> 4, mr = lane & 15u;
    const unsigned m0   = blockIdx.x * 32u;
    const unsigned rt   = wave >> 2;                  // row tile 0/1
    const unsigned c0   = (wave & 3u) * (unsigned)TPW;

    v8f acc[TPW];
    #pragma unroll
    for (int q = 0; q < TPW; ++q) acc[q] = (v8f){};

    const unsigned nk = K >> 5;

    auto load_tiles = [&](unsigned t, unsigned buf) {
        const unsigned k0 = t << 5;
        {   // A tile 32x32 f32 = 256 float4, one per thread (coalesced)
            unsigned r = tid >> 3, cq = (tid & 7u) * 4u;
            const vf4* ap = (const vf4*)(A + (m0 + r) * K + k0 + cq);
            vf4 v = NTA ? __builtin_nontemporal_load(ap) : *ap;
            unsigned ln = ((cq >> 3) & 1u) * 16u + (r & 15u);
            unsigned p  = (cq & 7u) | ((cq >> 4) << 3);
            *(vu2*)&Af[buf][r >> 4][ln][p] = pack4_bf16(v);
        }
        #pragma unroll
        for (int s = 0; s < NT / 2; ++s) {            // B: NT*128 float4
            unsigned j  = tid + 256u * (unsigned)s;
            unsigned nn = j >> 3, kq = (j & 7u) * 4u;
            vf4 v = *(const vf4*)(Wt + nn * K + k0 + kq);
            unsigned ct = nn >> 4;
            unsigned ln = ((kq >> 3) & 1u) * 16u + (nn & 15u);
            unsigned p  = (kq & 7u) | ((kq >> 4) << 3);
            *(vu2*)&Bf[buf][ct][ln][p] = pack4_bf16(v);
        }
    };

    load_tiles(0, 0);
    __syncthreads();
    for (unsigned t = 0; t < nk; ++t) {
        const unsigned buf = t & 1u;
        // 1) issue fragment ds_loads first (2x ds_load_b128 each)
        v16bf af = *(const v16bf*)&Af[buf][rt][lane][0];
        v16bf bv[TPW];
        #pragma unroll
        for (int q = 0; q < TPW; ++q)
            bv[q] = *(const v16bf*)&Bf[buf][c0 + q][lane][0];
        // 2) then prefetch next tile (global loads + ds_stores overlap WMMA)
        if (t + 1 < nk) load_tiles(t + 1, buf ^ 1u);
        // 3) matrix ops
        #pragma unroll
        for (int q = 0; q < TPW; ++q)
            acc[q] = __builtin_amdgcn_wmma_f32_16x16x32_bf16(
                         false, af, false, bv[q], (short)0, acc[q], false, false);
        __syncthreads();
    }

    // epilogue: D layout -> VGPR v holds row (v + 8*half), col = lane&15
    #pragma unroll
    for (int q = 0; q < TPW; ++q) {
        unsigned nc = (c0 + (unsigned)q) * 16u + mr;
        float bv2 = bias[nc];
        #pragma unroll
        for (int v = 0; v < 8; ++v) {
            unsigned mrow = m0 + rt * 16u + (unsigned)v + 8u * half;
            C[mrow * N + nc] = acc[q][v] + bv2;
        }
    }
}

// ---------------------------------------------------------------------------
// BatchNorm (training batch stats)
// ---------------------------------------------------------------------------
__global__ __launch_bounds__(256)
void bn_stats(const float* __restrict__ H, unsigned M, unsigned N,
              float* __restrict__ mu, float* __restrict__ rstd)
{
    __shared__ float s1[256], s2[256];
    const unsigned c = blockIdx.x;
    float sum = 0.f, sq = 0.f;
    for (unsigned r = threadIdx.x; r < M; r += 256u) {
        float v = H[r * N + c];
        sum += v; sq += v * v;
    }
    s1[threadIdx.x] = sum; s2[threadIdx.x] = sq;
    __syncthreads();
    for (int s = 128; s > 0; s >>= 1) {
        if ((int)threadIdx.x < s) {
            s1[threadIdx.x] += s1[threadIdx.x + s];
            s2[threadIdx.x] += s2[threadIdx.x + s];
        }
        __syncthreads();
    }
    if (threadIdx.x == 0) {
        float m   = s1[0] / (float)M;
        float var = s2[0] / (float)M - m * m;
        mu[c]   = m;
        rstd[c] = rsqrtf(var + 1e-5f);
    }
}

__global__ __launch_bounds__(256)
void bn_relu(float* __restrict__ H, unsigned M, unsigned N,
             const float* __restrict__ mu, const float* __restrict__ rstd,
             const float* __restrict__ g, const float* __restrict__ be)
{
    unsigned idx = blockIdx.x * 256u + threadIdx.x;
    if (idx < M * N) {
        unsigned c = idx % N;
        float v = (H[idx] - mu[c]) * rstd[c] * g[c] + be[c];
        H[idx] = v > 0.f ? v : 0.f;
    }
}

// ---------------------------------------------------------------------------
// Fused attention: msg = (adj==1 ? alpha : 0) @ h ; deg = rowsum(adj)
// out = deg!=0 ? msg*W/deg + h : 0.  Block = 32 rows x 64 cols, 8 waves,
// wave w -> row tile (w>>2), col tile (w&3). adj/alpha read exactly once.
// ---------------------------------------------------------------------------
__global__ __launch_bounds__(256)
void attn_fused(const int*   __restrict__ adj,
                const float* __restrict__ alpha,
                const float* __restrict__ Hf,     // [n,64]
                const float* __restrict__ Wscal,  // [1]
                float*       __restrict__ Out,    // [n,64]
                unsigned n)
{
    __shared__ __align__(32) bf16_t Af[2][2][32][16];
    __shared__ __align__(32) bf16_t Bf[2][4][32][16];
    __shared__ float degs[32];

    const unsigned tid  = threadIdx.x;
    const unsigned lane = tid & 31u, wave = tid >> 5;
    const unsigned half = lane >> 4, mr = lane & 15u;
    const unsigned i0 = blockIdx.x * 32u;
    const unsigned rt = wave >> 2, ct = wave & 3u;

    if (tid < 32) degs[tid] = 0.f;

    float degp = 0.f;
    v8f acc = {};

    auto load_tiles = [&](unsigned t, unsigned buf) {
        const unsigned k0 = t << 5;
        {   // masked A tile 32x32: one int4+float4 per thread (stream-once)
            unsigned r = tid >> 3, cq = (tid & 7u) * 4u;
            unsigned gi = (i0 + r) * n + k0 + cq;
            vi4 av = __builtin_nontemporal_load((const vi4*)(adj + gi));
            vf4 al = __builtin_nontemporal_load((const vf4*)(alpha + gi));
            vf4 m;
            m.x = (av.x == 1) ? al.x : 0.f;
            m.y = (av.y == 1) ? al.y : 0.f;
            m.z = (av.z == 1) ? al.z : 0.f;
            m.w = (av.w == 1) ? al.w : 0.f;
            degp += (float)(av.x + av.y + av.z + av.w);
            unsigned ln = ((cq >> 3) & 1u) * 16u + (r & 15u);
            unsigned p  = (cq & 7u) | ((cq >> 4) << 3);
            *(vu2*)&Af[buf][r >> 4][ln][p] = pack4_bf16(m);
        }
        #pragma unroll
        for (int s = 0; s < 2; ++s) {     // B tile 32x64 from h (row-major)
            unsigned nc = tid & 63u, kq = (tid >> 6) * 4u + (unsigned)s * 16u;
            vf4 v;
            v.x = Hf[(k0 + kq + 0u) * 64u + nc];
            v.y = Hf[(k0 + kq + 1u) * 64u + nc];
            v.z = Hf[(k0 + kq + 2u) * 64u + nc];
            v.w = Hf[(k0 + kq + 3u) * 64u + nc];
            unsigned ctb = nc >> 4;
            unsigned ln  = ((kq >> 3) & 1u) * 16u + (nc & 15u);
            unsigned p   = (kq & 7u) | ((kq >> 4) << 3);
            *(vu2*)&Bf[buf][ctb][ln][p] = pack4_bf16(v);
        }
    };

    load_tiles(0, 0);
    __syncthreads();
    const unsigned nk = n >> 5;
    for (unsigned t = 0; t < nk; ++t) {
        const unsigned buf = t & 1u;
        // fragment loads first, then prefetch, then WMMA (overlap)
        v16bf af = *(const v16bf*)&Af[buf][rt][lane][0];
        v16bf bv = *(const v16bf*)&Bf[buf][ct][lane][0];
        if (t + 1 < nk) load_tiles(t + 1, buf ^ 1u);
        acc = __builtin_amdgcn_wmma_f32_16x16x32_bf16(
                  false, af, false, bv, (short)0, acc, false, false);
        __syncthreads();
    }

    atomicAdd(&degs[tid >> 3], degp);     // ds_add_f32, once per thread
    __syncthreads();

    const float Wv = Wscal[0];
    #pragma unroll
    for (int v = 0; v < 8; ++v) {
        unsigned r  = rt * 16u + (unsigned)v + 8u * half;
        unsigned gi = i0 + r;
        float d = degs[r];
        unsigned c = ct * 16u + mr;
        float h = Hf[gi * 64u + c];
        Out[gi * 64u + c] = (d != 0.f) ? acc[v] * Wv / d + h : 0.f;
    }
}

// ---------------------------------------------------------------------------
// Classifier: deterministic two-stage reduction + softmax(2)
// ---------------------------------------------------------------------------
__global__ __launch_bounds__(256)
void cls_reduce(const float* __restrict__ feat,
                const float* __restrict__ w0, const float* __restrict__ w1,
                unsigned len, float* __restrict__ part)
{
    __shared__ float s0[256], s1[256];
    unsigned stride = gridDim.x * 256u;
    float p0 = 0.f, p1 = 0.f;
    for (unsigned i = blockIdx.x * 256u + threadIdx.x; i < len; i += stride) {
        float f = feat[i];
        p0 += f * w0[i];
        p1 += f * w1[i];
    }
    s0[threadIdx.x] = p0; s1[threadIdx.x] = p1;
    __syncthreads();
    for (int s = 128; s > 0; s >>= 1) {
        if ((int)threadIdx.x < s) {
            s0[threadIdx.x] += s0[threadIdx.x + s];
            s1[threadIdx.x] += s1[threadIdx.x + s];
        }
        __syncthreads();
    }
    if (threadIdx.x == 0) {
        part[2 * blockIdx.x]     = s0[0];
        part[2 * blockIdx.x + 1] = s1[0];
    }
}

__global__ __launch_bounds__(256)
void cls_final(const float* __restrict__ part, int nblk,
               const float* __restrict__ cls_b, float* __restrict__ out)
{
    __shared__ float s0[256], s1[256];
    float p0 = 0.f, p1 = 0.f;
    for (int i = threadIdx.x; i < nblk; i += 256) {
        p0 += part[2 * i];
        p1 += part[2 * i + 1];
    }
    s0[threadIdx.x] = p0; s1[threadIdx.x] = p1;
    __syncthreads();
    for (int s = 128; s > 0; s >>= 1) {
        if ((int)threadIdx.x < s) {
            s0[threadIdx.x] += s0[threadIdx.x + s];
            s1[threadIdx.x] += s1[threadIdx.x + s];
        }
        __syncthreads();
    }
    if (threadIdx.x == 0) {
        float l0 = s0[0] + cls_b[0];
        float l1 = s1[0] + cls_b[1];
        float m  = fmaxf(l0, l1);
        float e0 = expf(l0 - m), e1 = expf(l1 - m);
        float s  = e0 + e1;
        out[0] = e0 / s;
        out[1] = e1 / s;
    }
}

// ---------------------------------------------------------------------------
// Launch sequence
// ---------------------------------------------------------------------------
extern "C" void kernel_launch(void* const* d_in, const int* in_sizes, int n_in,
                              void* d_out, int out_size, void* d_ws, size_t ws_size,
                              hipStream_t stream)
{
    (void)in_sizes; (void)n_in; (void)out_size; (void)ws_size;

    const unsigned n = 8192;
    const float* x1     = (const float*)d_in[0];
    const float* x2     = (const float*)d_in[1];
    const int*   adj1   = (const int*)d_in[2];
    const int*   adj2   = (const int*)d_in[3];
    const float* p[2][12];
    for (int g = 0; g < 2; ++g)
        for (int j = 0; j < 12; ++j)
            p[g][j] = (const float*)d_in[4 + 12 * g + j];
    const float* Wsc    = (const float*)d_in[28];
    const float* alpha1 = (const float*)d_in[29];
    const float* cls_w  = (const float*)d_in[30];
    const float* cls_b  = (const float*)d_in[31];

    float* ws      = (float*)d_ws;
    float* hA      = ws;                                  // 8192*256
    float* hB      = ws + 2097152;                        // 8192*128
    float* hC[2]   = { ws + 3145728, ws + 3670016 };      // 8192*64 each
    float* nw      = ws + 4194304;                        // new1 || new2 contiguous
    float* stat_mu = ws + 5242880;                        // 256
    float* stat_rs = ws + 5243136;                        // 256
    float* part    = ws + 5243392;                        // 2*1024

    for (int g = 0; g < 2; ++g) {
        const float* X  = g ? x2 : x1;
        const int*   Aj = g ? adj2 : adj1;
        // layer 1: [8192,8192] x [8192,256]  (x streamed once, non-temporal)
        gemm_wmma<4, true><<<n / 32, 256, 0, stream>>>(X, p[g][0], p[g][1], hA, n, 256u, n);
        bn_stats<<<256, 256, 0, stream>>>(hA, n, 256u, stat_mu, stat_rs);
        bn_relu<<<(n * 256) / 256, 256, 0, stream>>>(hA, n, 256u, stat_mu, stat_rs, p[g][2], p[g][3]);
        // layer 2: [8192,256] x [256,128]
        gemm_wmma<2, false><<<n / 32, 256, 0, stream>>>(hA, p[g][4], p[g][5], hB, n, 128u, 256u);
        bn_stats<<<128, 256, 0, stream>>>(hB, n, 128u, stat_mu, stat_rs);
        bn_relu<<<(n * 128) / 256, 256, 0, stream>>>(hB, n, 128u, stat_mu, stat_rs, p[g][6], p[g][7]);
        // layer 3: [8192,128] x [128,64]
        gemm_wmma<1, false><<<n / 32, 256, 0, stream>>>(hB, p[g][8], p[g][9], hC[g], n, 64u, 128u);
        bn_stats<<<64, 256, 0, stream>>>(hC[g], n, 64u, stat_mu, stat_rs);
        bn_relu<<<(n * 64) / 256, 256, 0, stream>>>(hC[g], n, 64u, stat_mu, stat_rs, p[g][10], p[g][11]);
        // fused attention (msg WMMA + degree + epilogue)
        attn_fused<<<n / 32, 256, 0, stream>>>(Aj, alpha1, hC[g], Wsc, nw + (size_t)g * 524288, n);
    }

    cls_reduce<<<1024, 256, 0, stream>>>(nw, cls_w, cls_w + 1048576, 1048576u, part);
    cls_final<<<1, 256, 0, stream>>>(part, 1024, cls_b, (float*)d_out);
}
